// GaussianFusionNet_18202071400554
// MI455X (gfx1250) — compile-verified, matrix operand order checked by reading, not run on hardware
//
#include <hip/hip_runtime.h>
#include <hip/hip_bf16.h>
#include <math.h>

// ---------------------------------------------------------------------------
// GaussianFusionNet forward for MI455X (gfx1250, wave32, WMMA).
// B=4, N=16384, V=6, C=384, H=W=32.
// ---------------------------------------------------------------------------

#define BN   4
#define NN   16384
#define NV   6
#define NC   384
#define HW   1024      // 32*32

typedef __attribute__((ext_vector_type(16))) __bf16 v16bf;
typedef __attribute__((ext_vector_type(8)))  float  v8f;

// bf16 round-to-nearest-even
__device__ __forceinline__ unsigned short f2bf(float f) {
  unsigned int u = __float_as_uint(f);
  unsigned int r = (u + 0x7FFFu + ((u >> 16) & 1u)) >> 16;
  return (unsigned short)r;
}
__device__ __forceinline__ unsigned int pack_bf2(float lo, float hi) {
  return (unsigned int)f2bf(lo) | ((unsigned int)f2bf(hi) << 16);
}
__device__ __forceinline__ float gelu_exact(float x) {
  return 0.5f * x * (1.0f + erff(x * 0.7071067811865476f));
}
// Combine with the other 16-lane half: ds_swizzle SWAPX16 (xor 0x10, and 0x1f)
__device__ __forceinline__ float xhalf_sum(float x) {
  int o = __builtin_amdgcn_ds_swizzle(__float_as_int(x), 0x401F);
  return x + __int_as_float(o);
}
__device__ __forceinline__ float xhalf_max(float x) {
  int o = __builtin_amdgcn_ds_swizzle(__float_as_int(x), 0x401F);
  return fmaxf(x, __int_as_float(o));
}
__device__ __forceinline__ v8f vz8() {
  v8f r;
#pragma unroll
  for (int i = 0; i < 8; ++i) r[i] = 0.0f;
  return r;
}

union FragBF { v16bf v; unsigned int w[8]; uint4 u[2]; };

// ---------------------------------------------------------------------------
// Kernel T: transpose v_feats (BV,C,H,W) -> vt (BV,H*W,C), fp32.
// Makes per-tap channel gathers contiguous (float4 = 4 channels / load).
// ---------------------------------------------------------------------------
__global__ __launch_bounds__(256) void kT(const float* __restrict__ src,
                                          float* __restrict__ dst) {
  __shared__ float tile[32][33];
  int bx  = blockIdx.x;
  int bv  = bx / (12 * 32);
  int rem = bx % (12 * 32);
  int c0  = (rem / 32) * 32;
  int h0  = (rem % 32) * 32;
  int tid = threadIdx.x;
  int tx  = tid & 31, ty = tid >> 5;   // ty in 0..7
#pragma unroll
  for (int i = 0; i < 4; ++i) {
    int c = ty + 8 * i;
    tile[c][tx] = src[((size_t)(bv * NC + c0 + c)) * HW + h0 + tx];
  }
  __syncthreads();
#pragma unroll
  for (int i = 0; i < 4; ++i) {
    int h = ty + 8 * i;
    dst[((size_t)(bv * HW + h0 + h)) * NC + c0 + tx] = tile[tx][h];
  }
}

// ---------------------------------------------------------------------------
// Kernel A: projection + bilinear sampling + 384->64 (WMMA) + GELU +
//           64->32 (WMMA) + per-point LN + gated/masked accumulation.
// 256 threads = 8 waves, each wave owns 16 points. 128 points / block.
// Epilogue uses lane==point layout: no cross-lane shuffles except one
// SWAPX16 swizzle per statistic.
// ---------------------------------------------------------------------------
#define W1_STR 392   // padded bf16 row stride for bn_w1 [n=64][k=384]
#define W2_STR 72    // padded bf16 row stride for bn_w2 [n=32][k=64]
#define H_STR  72    // padded bf16 row stride for h [p=16][k=64]
#define A2_STR 36    // padded fp32 row stride for a2 staging [p=16][n=32]

__global__ __launch_bounds__(256) void kA(
    const float* __restrict__ xyz, const float* __restrict__ g14d,
    const float* __restrict__ vt, const float* __restrict__ ext,
    const float* __restrict__ intr, const float* __restrict__ offsets,
    const unsigned char* __restrict__ mask,
    const float* __restrict__ gate_w1, const float* __restrict__ gate_b1,
    const float* __restrict__ gate_w2, const float* __restrict__ gate_b2,
    const float* __restrict__ bn_w1, const float* __restrict__ bn_b1,
    const float* __restrict__ bn_w2, const float* __restrict__ bn_b2,
    const float* __restrict__ bn_ln_w, const float* __restrict__ bn_ln_b,
    float* __restrict__ fused) {
  __shared__ unsigned short w1s[64 * W1_STR];        // bn_w1^T as bf16 [n][k]
  __shared__ unsigned short w2s[32 * W2_STR];        // bn_w2^T as bf16 [n][k]
  __shared__ unsigned short hs[8][16 * H_STR];       // per-wave h staging (bf16)
  __shared__ float a2s_all[8][16 * A2_STR];          // per-wave a2 staging (fp32)

  int tid  = threadIdx.x;
  int lane = tid & 31;
  int wv   = tid >> 5;
  int bx   = blockIdx.x;
  int b    = bx >> 7;                 // 128 blocks per batch
  int n0   = (bx & 127) << 7;         // 128 points per block

  // Stage bn_w1 (384x64) transposed -> [n][k] bf16, padded stride.
  for (int i = tid; i < NC * 64; i += 256) {
    int k = i >> 6, n = i & 63;
    w1s[n * W1_STR + k] = f2bf(bn_w1[i]);
  }
  // Stage bn_w2 (64x32) transposed -> [n][k] bf16.
  for (int i = tid; i < 64 * 32; i += 256) {
    int k = i >> 5, n = i & 31;
    w2s[n * W2_STR + k] = f2bf(bn_w2[i]);
  }
  __syncthreads();

  int pl   = lane & 15;               // lane's point within the wave's 16
  int hi   = lane >> 4;               // half-wave (channel-group select)
  int coff = hi << 3;                 // 0 or 8 : A-frag K-group offset
  int pt   = n0 + (wv << 4) + pl;     // point within batch
  size_t gp = (size_t)b * NN + pt;

  // per-point constants (lanes l and l+16 redundantly hold point pl)
  float x3 = xyz[gp * 3 + 0], y3 = xyz[gp * 3 + 1], z3 = xyz[gp * 3 + 2];
  float off0 = offsets[gp * 2 + 0], off1 = offsets[gp * 2 + 1];
  float maskv = mask[gp] ? 1.0f : 0.0f;

  // PhysicsGatedProjection gate: sigmoid(gelu([s3,op] W1 + b1) W2 + b2)
  float fg0 = fminf(fmaxf(g14d[gp * 14 + 4], 0.01f), 10.0f);
  float fg1 = fminf(fmaxf(g14d[gp * 14 + 5], 0.01f), 10.0f);
  float fg2 = fminf(fmaxf(g14d[gp * 14 + 6], 0.01f), 10.0f);
  float fg3 = fminf(fmaxf(g14d[gp * 14 + 11], 0.0f), 1.0f);
  float gacc = gate_b2[0];
#pragma unroll
  for (int j = 0; j < 16; ++j) {
    float h = gate_b1[j] + fg0 * gate_w1[j] + fg1 * gate_w1[16 + j] +
              fg2 * gate_w1[32 + j] + fg3 * gate_w1[48 + j];
    gacc += gelu_exact(h) * gate_w2[j];
  }
  float gatev = 1.0f / (1.0f + expf(-gacc));

  float b1t[4], b2t[2];
#pragma unroll
  for (int t = 0; t < 4; ++t) b1t[t] = bn_b1[16 * t + pl];
#pragma unroll
  for (int t = 0; t < 2; ++t) b2t[t] = bn_b2[16 * t + pl];

  // LN params for this lane's 16 output columns (hi*16 .. hi*16+15)
  float lnw[16], lnb[16];
#pragma unroll
  for (int j = 0; j < 16; ++j) {
    lnw[j] = bn_ln_w[hi * 16 + j];
    lnb[j] = bn_ln_b[hi * 16 + j];
  }

  float sa[16];                        // per-point fused accumulator (16 cols)
#pragma unroll
  for (int j = 0; j < 16; ++j) sa[j] = 0.0f;
  float summn = 0.0f;

  unsigned short* hsw = hs[wv];
  float* a2s = a2s_all[wv];

  for (int v = 0; v < NV; ++v) {
    const float* E  = ext  + (size_t)(b * NV + v) * 12;
    const float* Ki = intr + (size_t)(b * NV + v) * 9;
    float cx = E[0] * x3 + E[1] * y3 + E[2]  * z3 + E[3];
    float cy = E[4] * x3 + E[5] * y3 + E[6]  * z3 + E[7];
    float cz = E[8] * x3 + E[9] * y3 + E[10] * z3 + E[11];
    float ux = Ki[0] * cx + Ki[1] * cy + Ki[2] * cz;
    float uy = Ki[3] * cx + Ki[4] * cy + Ki[5] * cz;
    float uz = Ki[6] * cx + Ki[7] * cy + Ki[8] * cz;
    float depth = fmaxf(uz, 1e-6f);
    float gxv = 2.0f * (ux / depth) * (1.0f / 448.0f) - 1.0f + off0;
    float gyv = 2.0f * (uy / depth) * (1.0f / 448.0f) - 1.0f + off1;
    float validf =
        ((depth > 0.1f) && (fabsf(gxv) <= 1.0f) && (fabsf(gyv) <= 1.0f))
            ? maskv : 0.0f;

    // grid_sample bilinear, align_corners=False, zeros padding
    float xs  = ((gxv + 1.0f) * 32.0f - 1.0f) * 0.5f;
    float ys  = ((gyv + 1.0f) * 32.0f - 1.0f) * 0.5f;
    float x0f = floorf(xs), y0f = floorf(ys);
    float wx1 = xs - x0f,  wy1 = ys - y0f;
    int x0i = (int)x0f, y0i = (int)y0f;
    const float* vtb = vt + (size_t)(b * NV + v) * HW * NC;

    const float* tp[4];
    float tw[4];
    {
      int xi[2] = {x0i, x0i + 1};
      int yi[2] = {y0i, y0i + 1};
      float wxv[2] = {1.0f - wx1, wx1};
      float wyv[2] = {1.0f - wy1, wy1};
      int idx = 0;
#pragma unroll
      for (int jy = 0; jy < 2; ++jy)
#pragma unroll
        for (int jx = 0; jx < 2; ++jx) {
          bool tv = (xi[jx] >= 0) && (xi[jx] < 32) && (yi[jy] >= 0) && (yi[jy] < 32);
          int cxi = min(max(xi[jx], 0), 31);
          int cyi = min(max(yi[jy], 0), 31);
          tw[idx] = tv ? wxv[jx] * wyv[jy] : 0.0f;
          tp[idx] = vtb + (size_t)(cyi * 32 + cxi) * NC;
          ++idx;
        }
    }
    const float* tpa = tp[0]; const float* tpb = tp[1];
    const float* tpc = tp[2]; const float* tpd = tp[3];
    float wa = tw[0], wb = tw[1], wc = tw[2], wd = tw[3];

    // ---- 384 -> 64 via WMMA bf16 : M=16 pts, N=64 (4 tiles), K=12x32 ----
    v8f acc0 = vz8(), acc1 = vz8(), acc2_ = vz8(), acc3 = vz8();
    for (int kc = 0; kc < 12; ++kc) {
      int c1 = kc * 32 + coff;          // lane's channel groups: c1..c1+7, c1+16..c1+23
      float s[16];
#pragma unroll
      for (int g = 0; g < 4; ++g) {
        int cg = (g < 2) ? (c1 + 4 * g) : (c1 + 8 + 4 * g);
        float4 ta = *(const float4*)(tpa + cg);
        float4 tb_ = *(const float4*)(tpb + cg);
        float4 tc_ = *(const float4*)(tpc + cg);
        float4 td_ = *(const float4*)(tpd + cg);
        s[4 * g + 0] = wa * ta.x + wb * tb_.x + wc * tc_.x + wd * td_.x;
        s[4 * g + 1] = wa * ta.y + wb * tb_.y + wc * tc_.y + wd * td_.y;
        s[4 * g + 2] = wa * ta.z + wb * tb_.z + wc * tc_.z + wd * td_.z;
        s[4 * g + 3] = wa * ta.w + wb * tb_.w + wc * tc_.w + wd * td_.w;
      }
      FragBF af;
#pragma unroll
      for (int j = 0; j < 8; ++j) af.w[j] = pack_bf2(s[2 * j], s[2 * j + 1]);
#pragma unroll
      for (int t = 0; t < 4; ++t) {
        FragBF bf_;
        const unsigned short* wr = &w1s[(16 * t + pl) * W1_STR + kc * 32 + coff];
        bf_.u[0] = *(const uint4*)(wr);
        bf_.u[1] = *(const uint4*)(wr + 16);
        v8f* ac = (t == 0) ? &acc0 : (t == 1) ? &acc1 : (t == 2) ? &acc2_ : &acc3;
        *ac = __builtin_amdgcn_wmma_f32_16x16x32_bf16(
            false, af.v, false, bf_.v, (short)0, *ac, false, false);
      }
    }

    // h = gelu(acc + b1)  -> per-wave LDS (C-layout -> row-major bf16)
#pragma unroll
    for (int t = 0; t < 4; ++t) {
      v8f* ac = (t == 0) ? &acc0 : (t == 1) ? &acc1 : (t == 2) ? &acc2_ : &acc3;
#pragma unroll
      for (int r = 0; r < 8; ++r) {
        float hv = gelu_exact((*ac)[r] + b1t[t]);
        int p = r + 8 * hi;
        hsw[p * H_STR + 16 * t + pl] = f2bf(hv);
      }
    }

    // ---- 64 -> 32 via WMMA bf16 : K=2x32, N=32 (2 tiles) ----
    v8f a20 = vz8(), a21 = vz8();
#pragma unroll
    for (int kc2 = 0; kc2 < 2; ++kc2) {
      FragBF af2;
      const unsigned short* hr = &hsw[pl * H_STR + kc2 * 32 + coff];
      af2.u[0] = *(const uint4*)(hr);
      af2.u[1] = *(const uint4*)(hr + 16);
#pragma unroll
      for (int t = 0; t < 2; ++t) {
        FragBF bf_;
        const unsigned short* wr = &w2s[(16 * t + pl) * W2_STR + kc2 * 32 + coff];
        bf_.u[0] = *(const uint4*)(wr);
        bf_.u[1] = *(const uint4*)(wr + 16);
        v8f* ac = (t == 0) ? &a20 : &a21;
        *ac = __builtin_amdgcn_wmma_f32_16x16x32_bf16(
            false, af2.v, false, bf_.v, (short)0, *ac, false, false);
      }
    }

    // ---- bias add, C-layout -> per-wave fp32 staging [p][A2_STR] ----
#pragma unroll
    for (int r = 0; r < 8; ++r) {
      int p = r + 8 * hi;
      a2s[p * A2_STR + pl]      = a20[r] + b2t[0];
      a2s[p * A2_STR + 16 + pl] = a21[r] + b2t[1];
    }
    // wave-private region: DS ops of one wave are in order, no barrier needed.

    // ---- per-point LayerNorm(32): lane -> point pl, half -> 16 cols ----
    const float* rowp = &a2s[pl * A2_STR + hi * 16];
    float4 q0 = *(const float4*)(rowp + 0);
    float4 q1 = *(const float4*)(rowp + 4);
    float4 q2 = *(const float4*)(rowp + 8);
    float4 q3 = *(const float4*)(rowp + 12);
    float xv[16] = {q0.x, q0.y, q0.z, q0.w, q1.x, q1.y, q1.z, q1.w,
                    q2.x, q2.y, q2.z, q2.w, q3.x, q3.y, q3.z, q3.w};
    float s_ = 0.0f, sq_ = 0.0f;
#pragma unroll
    for (int j = 0; j < 16; ++j) { s_ += xv[j]; sq_ += xv[j] * xv[j]; }
    s_  = xhalf_sum(s_);
    sq_ = xhalf_sum(sq_);
    float mean = s_ * (1.0f / 32.0f);
    float var  = fmaxf(sq_ * (1.0f / 32.0f) - mean * mean, 0.0f);
    float inv  = rsqrtf(var + 1e-5f);
    float wgt  = gatev * validf;
#pragma unroll
    for (int j = 0; j < 16; ++j)
      sa[j] += ((xv[j] - mean) * inv * lnw[j] + lnb[j]) * wgt;
    summn += validf;
  }

  // fused = sum_f / (sum_m + 1e-6): lane stores its point's 16 contiguous cols
  float invm = 1.0f / (summn + 1e-6f);
  float* fb = fused + ((size_t)b * NN + n0 + wv * 16 + pl) * 32 + hi * 16;
#pragma unroll
  for (int g = 0; g < 4; ++g) {
    float4 o;
    o.x = sa[4 * g + 0] * invm;
    o.y = sa[4 * g + 1] * invm;
    o.z = sa[4 * g + 2] * invm;
    o.w = sa[4 * g + 3] * invm;
    *(float4*)(fb + 4 * g) = o;
  }
}

// ---------------------------------------------------------------------------
// Kernel B: act = softmax(LN(fused@cw)) * mask (WMMA, per-point epilogue),
//           then partial vlad[d][k] = sum_n act[n][k]*fused[n][d], a_sum[k].
// 32 blocks: b = bx>>3, part = bx&7, 2048 points each.
// ---------------------------------------------------------------------------
#define AS_STR 68   // padded fp32 row stride for act [p=128][k=64]

__global__ __launch_bounds__(256) void kB(
    const float* __restrict__ fused, const unsigned char* __restrict__ mask,
    const float* __restrict__ cw, const float* __restrict__ ln1w,
    const float* __restrict__ ln1b, float* __restrict__ pv,
    float* __restrict__ pa) {
  __shared__ float fs[128 * 33];             // fused fp32 [p][d]
  __shared__ unsigned short fbf[128 * 40];   // fused bf16 [p][d]
  __shared__ float as_[128 * AS_STR];        // logits -> act fp32 [p][k]
  __shared__ float ms[128];                  // mask

  int tid  = threadIdx.x;
  int lane = tid & 31;
  int wv   = tid >> 5;
  int bx   = blockIdx.x;
  int b    = bx >> 3;
  int part = bx & 7;
  int pl   = lane & 15;
  int hi   = lane >> 4;
  int koff = hi << 3;

  // cw (32x64) resident as 4 B-fragments (bf16)
  FragBF cbf[4];
#pragma unroll
  for (int t = 0; t < 4; ++t) {
    int n = 16 * t + pl;
#pragma unroll
    for (int j = 0; j < 4; ++j) {
      cbf[t].w[j] = pack_bf2(cw[(koff + 2 * j) * 64 + n],
                             cw[(koff + 2 * j + 1) * 64 + n]);
      cbf[t].w[4 + j] = pack_bf2(cw[(koff + 16 + 2 * j) * 64 + n],
                                 cw[(koff + 17 + 2 * j) * 64 + n]);
    }
  }
  // LN params for this lane's 32 columns (hi*32 .. hi*32+31)
  float lw[32], lb[32];
#pragma unroll
  for (int j = 0; j < 32; ++j) {
    lw[j] = ln1w[hi * 32 + j];
    lb[j] = ln1b[hi * 32 + j];
  }

  int k_t    = tid & 63;
  int d_base = (tid >> 6) * 8;
  float accv[8];
#pragma unroll
  for (int j = 0; j < 8; ++j) accv[j] = 0.0f;
  float asum = 0.0f;

  for (int st = 0; st < 16; ++st) {
    int p0 = part * 2048 + st * 128;
    __syncthreads();  // protect previous tile's fs/as_ reads
    for (int i = tid; i < 128 * 32; i += 256) {
      int p = i >> 5, d = i & 31;
      float fv = fused[((size_t)b * NN + p0 + p) * 32 + d];
      fs[p * 33 + d]  = fv;
      fbf[p * 40 + d] = f2bf(fv);
    }
    if (tid < 128) ms[tid] = mask[(size_t)b * NN + p0 + tid] ? 1.0f : 0.0f;
    __syncthreads();

    // logits = fused@cw : M=16 pts/wave, K=32, N=64 (4 tiles)
    FragBF af;
    const unsigned short* fr = &fbf[(wv * 16 + pl) * 40 + koff];
    af.u[0] = *(const uint4*)(fr);
    af.u[1] = *(const uint4*)(fr + 16);
    v8f lg0 = vz8(), lg1 = vz8(), lg2 = vz8(), lg3 = vz8();
    lg0 = __builtin_amdgcn_wmma_f32_16x16x32_bf16(false, af.v, false, cbf[0].v, (short)0, lg0, false, false);
    lg1 = __builtin_amdgcn_wmma_f32_16x16x32_bf16(false, af.v, false, cbf[1].v, (short)0, lg1, false, false);
    lg2 = __builtin_amdgcn_wmma_f32_16x16x32_bf16(false, af.v, false, cbf[2].v, (short)0, lg2, false, false);
    lg3 = __builtin_amdgcn_wmma_f32_16x16x32_bf16(false, af.v, false, cbf[3].v, (short)0, lg3, false, false);

    // C-layout -> as_ rows (each wave writes only its own 16 rows)
#pragma unroll
    for (int r = 0; r < 8; ++r) {
      int p = wv * 16 + r + 8 * hi;
      as_[p * AS_STR + pl]      = lg0[r];
      as_[p * AS_STR + 16 + pl] = lg1[r];
      as_[p * AS_STR + 32 + pl] = lg2[r];
      as_[p * AS_STR + 48 + pl] = lg3[r];
    }
    // per-point LN(64) + softmax(64): lane -> point wv*16+pl, half -> 32 cols
    {
      int p = wv * 16 + pl;
      float* rowp = &as_[p * AS_STR + hi * 32];
      float xv[32];
#pragma unroll
      for (int g = 0; g < 8; ++g) {
        float4 q = *(const float4*)(rowp + 4 * g);
        xv[4 * g + 0] = q.x; xv[4 * g + 1] = q.y;
        xv[4 * g + 2] = q.z; xv[4 * g + 3] = q.w;
      }
      float s_ = 0.0f, sq_ = 0.0f;
#pragma unroll
      for (int j = 0; j < 32; ++j) { s_ += xv[j]; sq_ += xv[j] * xv[j]; }
      s_  = xhalf_sum(s_);
      sq_ = xhalf_sum(sq_);
      float mean = s_ * (1.0f / 64.0f);
      float var  = fmaxf(sq_ * (1.0f / 64.0f) - mean * mean, 0.0f);
      float inv  = rsqrtf(var + 1e-5f);
      float mx = -3.4e38f;
#pragma unroll
      for (int j = 0; j < 32; ++j) {
        xv[j] = (xv[j] - mean) * inv * lw[j] + lb[j];
        mx = fmaxf(mx, xv[j]);
      }
      mx = xhalf_max(mx);
      float es = 0.0f;
#pragma unroll
      for (int j = 0; j < 32; ++j) { xv[j] = expf(xv[j] - mx); es += xv[j]; }
      es = xhalf_sum(es);
      float sc = ms[p] / es;
#pragma unroll
      for (int g = 0; g < 8; ++g) {
        float4 q;
        q.x = xv[4 * g + 0] * sc; q.y = xv[4 * g + 1] * sc;
        q.z = xv[4 * g + 2] * sc; q.w = xv[4 * g + 3] * sc;
        *(float4*)(rowp + 4 * g) = q;
      }
    }
    __syncthreads();

    // partial vlad += fused^T act  (register-blocked outer product)
    for (int p = 0; p < 128; ++p) {
      float a = as_[p * AS_STR + k_t];
      if (tid < 64) asum += a;
#pragma unroll
      for (int j = 0; j < 8; ++j)
        accv[j] += fs[p * 33 + d_base + j] * a;
    }
  }

  float* pvp = pv + (size_t)bx * 2048;
#pragma unroll
  for (int j = 0; j < 8; ++j) pvp[(d_base + j) * 64 + k_t] = accv[j];
  if (tid < 64) pa[bx * 64 + tid] = asum;
}

// ---------------------------------------------------------------------------
// Kernel C: reduce partials, vlad = raw - a_sum*cw2, intra+global L2 norms,
//           out = vlad@hw, gates = sigmoid(LN(out@gw)), result = out*gates.
// One block per batch, 256 threads.
// ---------------------------------------------------------------------------
__global__ __launch_bounds__(256) void kC(
    const float* __restrict__ pv, const float* __restrict__ pa,
    const float* __restrict__ cw2, const float* __restrict__ hwm,
    const float* __restrict__ gw, const float* __restrict__ glnw,
    const float* __restrict__ glnb, float* __restrict__ outp) {
  __shared__ float vl[32 * 65];
  __shared__ float aks[64];
  __shared__ float nk[64];
  __shared__ float outs[256];
  __shared__ float red[256];
  int b = blockIdx.x, tid = threadIdx.x;

  if (tid < 64) {
    float s = 0.0f;
    for (int part = 0; part < 8; ++part) s += pa[(b * 8 + part) * 64 + tid];
    aks[tid] = s;
  }
  __syncthreads();
  for (int i = tid; i < 2048; i += 256) {
    int d = i >> 6, k = i & 63;
    float s = 0.0f;
    for (int part = 0; part < 8; ++part) s += pv[(size_t)(b * 8 + part) * 2048 + i];
    s -= aks[k] * cw2[i];
    vl[d * 65 + k] = s;
  }
  __syncthreads();
  if (tid < 64) {
    float s = 0.0f;
    for (int d = 0; d < 32; ++d) { float v = vl[d * 65 + tid]; s += v * v; }
    nk[tid] = 1.0f / fmaxf(sqrtf(s), 1e-12f);
  }
  __syncthreads();
  float tot = 0.0f;
  for (int i = tid; i < 2048; i += 256) {
    int d = i >> 6, k = i & 63;
    float v = vl[d * 65 + k] * nk[k];
    vl[d * 65 + k] = v;
    tot += v * v;
  }
  red[tid] = tot;
  __syncthreads();
  for (int s = 128; s > 0; s >>= 1) {
    if (tid < s) red[tid] += red[tid + s];
    __syncthreads();
  }
  float ginv = 1.0f / fmaxf(sqrtf(red[0]), 1e-12f);
  __syncthreads();

  // out[j] = (vlad_flat . hw[:,j]) * ginv
  float oj = 0.0f;
  for (int i = 0; i < 2048; ++i)
    oj += vl[(i >> 6) * 65 + (i & 63)] * hwm[(size_t)i * 256 + tid];
  oj *= ginv;
  outs[tid] = oj;
  __syncthreads();

  float gj = 0.0f;
  for (int i = 0; i < 256; ++i) gj += outs[i] * gw[i * 256 + tid];
  red[tid] = gj;
  __syncthreads();
  for (int s = 128; s > 0; s >>= 1) {
    if (tid < s) red[tid] += red[tid + s];
    __syncthreads();
  }
  float mean = red[0] * (1.0f / 256.0f);
  __syncthreads();
  float dg = gj - mean;
  red[tid] = dg * dg;
  __syncthreads();
  for (int s = 128; s > 0; s >>= 1) {
    if (tid < s) red[tid] += red[tid + s];
    __syncthreads();
  }
  float var = red[0] * (1.0f / 256.0f);
  float g = dg * rsqrtf(var + 1e-5f) * glnw[tid] + glnb[tid];
  float gate = 1.0f / (1.0f + expf(-g));
  outp[b * 256 + tid] = oj * gate;
}

// ---------------------------------------------------------------------------
extern "C" void kernel_launch(void* const* d_in, const int* in_sizes, int n_in,
                              void* d_out, int out_size, void* d_ws, size_t ws_size,
                              hipStream_t stream) {
  const float* xyz      = (const float*)d_in[0];
  const float* g14d     = (const float*)d_in[1];
  const float* v_feats  = (const float*)d_in[2];
  const float* ext      = (const float*)d_in[3];
  const float* intr     = (const float*)d_in[4];
  const float* offsets  = (const float*)d_in[5];
  const unsigned char* mask = (const unsigned char*)d_in[6];
  const float* gate_w1  = (const float*)d_in[7];
  const float* gate_b1  = (const float*)d_in[8];
  const float* gate_w2  = (const float*)d_in[9];
  const float* gate_b2  = (const float*)d_in[10];
  const float* bn_w1    = (const float*)d_in[11];
  const float* bn_b1    = (const float*)d_in[12];
  const float* bn_w2    = (const float*)d_in[13];
  const float* bn_b2    = (const float*)d_in[14];
  const float* bn_ln_w  = (const float*)d_in[15];
  const float* bn_ln_b  = (const float*)d_in[16];
  const float* cw       = (const float*)d_in[17];
  const float* cw2      = (const float*)d_in[18];
  const float* hwm      = (const float*)d_in[19];
  const float* ln1_w    = (const float*)d_in[20];
  const float* ln1_b    = (const float*)d_in[21];
  const float* gw       = (const float*)d_in[22];
  const float* gln_w    = (const float*)d_in[23];
  const float* gln_b    = (const float*)d_in[24];
  float* outp = (float*)d_out;

  // Workspace layout (fully rewritten every call; ~46.4 MB):
  float* vt     = (float*)d_ws;                         // (BV,H*W,C) fp32
  float* fusedb = vt + (size_t)BN * NV * HW * NC;       // (B,N,32)
  float* pv     = fusedb + (size_t)BN * NN * 32;        // 32 x (32*64)
  float* pa     = pv + (size_t)32 * 2048;               // 32 x 64

  kT<<<BN * NV * 12 * 32, 256, 0, stream>>>(v_feats, vt);
  kA<<<(BN * NN) / 128, 256, 0, stream>>>(
      xyz, g14d, vt, ext, intr, offsets, mask,
      gate_w1, gate_b1, gate_w2, gate_b2,
      bn_w1, bn_b1, bn_w2, bn_b2, bn_ln_w, bn_ln_b, fusedb);
  kB<<<BN * 8, 256, 0, stream>>>(fusedb, mask, cw, ln1_w, ln1_b, pv, pa);
  kC<<<BN, 256, 0, stream>>>(pv, pa, cw2, hwm, gw, gln_w, gln_b, outp);
}
